// GNNEncoder_43190191129139
// MI455X (gfx1250) — matrix operand order repaired and checked by reading, not compiled, not run on hardware
//
#include <hip/hip_runtime.h>
#include <math.h>
#include <stdint.h>

// ---------------------------------------------------------------------------
// GNN GATv2 encoder for MI455X (gfx1250, wave32, WMMA, async global->LDS)
// ---------------------------------------------------------------------------

#define HID   128
#define NHEAD 4
#define NW_N  50000
#define NI_N  100000
#define E_WW  300000
#define E_II  600000
#define E_IW  300000
#define E_WI  300000
#define LRELU_S 0.2f
#define LN_EPS  1e-5f
#define WPAD  136   // padded K stride (bf16 elems); mult of 8 -> 16B frag align

typedef __attribute__((ext_vector_type(8)))  __bf16 v8bf;
typedef __attribute__((ext_vector_type(16))) __bf16 v16bf;
typedef __attribute__((ext_vector_type(8)))  float  v8f;
typedef __attribute__((ext_vector_type(4)))  int    v4i;

#define AS1 __attribute__((address_space(1)))
#define AS3 __attribute__((address_space(3)))

__device__ inline unsigned short f2bfbits(float f) {
    unsigned u = __float_as_uint(f);
    u += 0x7FFFu + ((u >> 16) & 1u);
    return (unsigned short)(u >> 16);
}
__device__ inline __bf16 f2bf(float f) {
    union { unsigned short s; __bf16 b; } cvt; cvt.s = f2bfbits(f); return cvt.b;
}

__device__ inline float atomicMaxF32(float* addr, float value) {
    if (value >= 0.0f) {
        int old = atomicMax((int*)addr, __float_as_int(value));
        return __int_as_float(old);
    } else {
        unsigned old = atomicMin((unsigned*)addr, (unsigned)__float_as_int(value));
        return __uint_as_float((int)old);
    }
}

// ---------------------------------------------------------------------------
// One-time weight conversion: W[128,128] f32 -> Wt[nCol][k] bf16, padded WPAD
// ---------------------------------------------------------------------------
__global__ void convert_w(const float* __restrict__ W, unsigned short* __restrict__ Wt)
{
    int i = blockIdx.x * blockDim.x + threadIdx.x;
    if (i >= 128 * WPAD) return;
    int nc = i / WPAD, k = i % WPAD;
    Wt[i] = (k < 128) ? f2bfbits(W[k * 128 + nc]) : (unsigned short)0;
}

// ---------------------------------------------------------------------------
// Dense GEMM: Y[n,128] = X[n,128] @ W[128,128] + bias   (bf16 WMMA, f32 acc)
// Block = 256 thr (8 waves). Each wave -> 16-row stripe: 8 N-tiles x 4 K-steps
// of v_wmma_f32_16x16x32_bf16. Pre-converted bf16 W^T staged to LDS via
// async global->LDS copies when available.
// ---------------------------------------------------------------------------
__global__ __launch_bounds__(256) void gemm128_bf16(
    const float* __restrict__ X, const unsigned short* __restrict__ Wt,
    const float* __restrict__ bias, float* __restrict__ Y, int n)
{
    __shared__ __align__(16) __bf16 sW[128 * WPAD];
    const int tid  = threadIdx.x;
    const int lane = tid & 31;
    const int wave = tid >> 5;
    const int half = lane >> 4;
    const int l15  = lane & 15;

    // ---- stage pre-converted bf16 W^T into LDS (34816 B = 2176 x b128) ----
    const int CH = (128 * WPAD * 2) / 16;
#if __has_builtin(__builtin_amdgcn_global_load_async_to_lds_b128)
    for (int i = tid; i < CH; i += 256) {
        AS1 v4i* g = (AS1 v4i*)(uintptr_t)((const char*)Wt + (size_t)i * 16);
        AS3 v4i* l = (AS3 v4i*)(unsigned)(uintptr_t)((char*)sW + (size_t)i * 16);
        __builtin_amdgcn_global_load_async_to_lds_b128(g, l, 0, 0);
    }
  #if __has_builtin(__builtin_amdgcn_s_wait_asynccnt)
    __builtin_amdgcn_s_wait_asynccnt(0);
  #else
    asm volatile("s_wait_asynccnt 0x0" ::: "memory");
  #endif
#else
    {
        const uint4* src = (const uint4*)Wt;
        uint4* dst = (uint4*)sW;
        for (int i = tid; i < CH; i += 256) dst[i] = src[i];
    }
#endif
    __syncthreads();

    const int rowBase = blockIdx.x * 128;
    int gr = rowBase + wave * 16 + l15;
    if (gr >= n) gr = n - 1;                    // clamp; stores guarded below
    const float* xrow = X + (size_t)gr * HID;

    v8f acc[8];
    #pragma unroll
    for (int t = 0; t < 8; ++t) {
        float bv = bias[t * 16 + l15];
        #pragma unroll
        for (int j = 0; j < 8; ++j) acc[t][j] = bv;
    }

    #pragma unroll
    for (int s = 0; s < 4; ++s) {
        // A fragment: lo8 = K 32s+8*half.. , hi8 = K 32s+16+8*half..
        const int kb = 32 * s + 8 * half;
        float4 f0 = *(const float4*)(xrow + kb);
        float4 f1 = *(const float4*)(xrow + kb + 4);
        float4 g0 = *(const float4*)(xrow + kb + 16);
        float4 g1 = *(const float4*)(xrow + kb + 20);
        v16bf a;
        a[0]=f2bf(f0.x); a[1]=f2bf(f0.y); a[2]=f2bf(f0.z); a[3]=f2bf(f0.w);
        a[4]=f2bf(f1.x); a[5]=f2bf(f1.y); a[6]=f2bf(f1.z); a[7]=f2bf(f1.w);
        a[8]=f2bf(g0.x); a[9]=f2bf(g0.y); a[10]=f2bf(g0.z); a[11]=f2bf(g0.w);
        a[12]=f2bf(g1.x); a[13]=f2bf(g1.y); a[14]=f2bf(g1.z); a[15]=f2bf(g1.w);

        // Preload all 8 B fragments, then issue 8 WMMAs (hide ds latency)
        v16bf bf[8];
        #pragma unroll
        for (int t = 0; t < 8; ++t) {
            const __bf16* bp = &sW[(t * 16 + l15) * WPAD + 32 * s + 8 * half];
            v8bf blo = *(const v8bf*)bp;
            v8bf bhi = *(const v8bf*)(bp + 16);
            #pragma unroll
            for (int i = 0; i < 8; ++i) { bf[t][i] = blo[i]; bf[t][i + 8] = bhi[i]; }
        }
        #pragma unroll
        for (int t = 0; t < 8; ++t)
            acc[t] = __builtin_amdgcn_wmma_f32_16x16x32_bf16(
                false, a, false, bf[t], (short)0, acc[t], false, false);
    }

    #pragma unroll
    for (int t = 0; t < 8; ++t) {
        int col = t * 16 + l15;
        #pragma unroll
        for (int j = 0; j < 8; ++j) {
            int r = rowBase + wave * 16 + half * 8 + j;
            if (r < n) Y[(size_t)r * HID + col] = acc[t][j];
        }
    }
}

// ---------------------------------------------------------------------------
// Input projections
// ---------------------------------------------------------------------------
__global__ __launch_bounds__(128) void proj_wifi(
    const float* __restrict__ X, const float* __restrict__ W,
    const float* __restrict__ b, const float* __restrict__ pos,
    const int* __restrict__ tim, float* __restrict__ Y)
{
    __shared__ float sx[64];
    const int nid = blockIdx.x, c = threadIdx.x;
    if (c < 64) sx[c] = X[(size_t)nid * 64 + c];
    __syncthreads();
    float acc = b[c] + pos[(size_t)tim[nid] * HID + c];
    #pragma unroll
    for (int k = 0; k < 64; ++k) acc += sx[k] * W[k * HID + c];
    Y[(size_t)nid * HID + c] = acc;
}

__global__ __launch_bounds__(128) void proj_imu(
    const float* __restrict__ X, const float* __restrict__ W,
    const float* __restrict__ b, const float* __restrict__ pos,
    const int* __restrict__ tim, float* __restrict__ Y)
{
    const int nid = blockIdx.x, c = threadIdx.x;
    float acc = b[c] + pos[(size_t)tim[nid] * HID + c];
    #pragma unroll
    for (int k = 0; k < 6; ++k) acc += X[(size_t)nid * 6 + k] * W[k * HID + c];
    Y[(size_t)nid * HID + c] = acc;
}

// ---------------------------------------------------------------------------
// Init helpers
// ---------------------------------------------------------------------------
__global__ void init_acc(float* __restrict__ o, const float* __restrict__ b1,
                         const float* __restrict__ b2, int total)
{
    for (int i = blockIdx.x * blockDim.x + threadIdx.x; i < total;
         i += gridDim.x * blockDim.x)
        o[i] = b1[i & 127] + b2[i & 127];
}

__global__ void init_ms(float* __restrict__ m, float* __restrict__ s, int total)
{
    for (int i = blockIdx.x * blockDim.x + threadIdx.x; i < total;
         i += gridDim.x * blockDim.x) { m[i] = -INFINITY; s[i] = 0.0f; }
}

// ---------------------------------------------------------------------------
// Edge pass 1: attention logits + segment max (wave per edge)
// ---------------------------------------------------------------------------
__global__ __launch_bounds__(256) void edge_attn(
    const int* __restrict__ src, const int* __restrict__ dst,
    const float* __restrict__ eattr, int de,
    const float* __restrict__ xl, const float* __restrict__ xr,
    const float* __restrict__ We, const float* __restrict__ att,
    float* __restrict__ a_out, float* __restrict__ m, int E)
{
    const int lane = threadIdx.x & 31;
    const int wid  = blockIdx.x * 8 + (threadIdx.x >> 5);
    const int nw   = gridDim.x * 8;
    const int c    = lane * 4;
    const float a0 = att[c], a1 = att[c + 1], a2 = att[c + 2], a3 = att[c + 3];
    const float w0 = We[c], w1 = We[c + 1], w2 = We[c + 2], w3 = We[c + 3];

    for (int e = wid; e < E; e += nw) {
        int s = src[e], d = dst[e];
        float4 vl = *(const float4*)(xl + (size_t)s * HID + c);
        float4 vr = *(const float4*)(xr + (size_t)d * HID + c);
        float e0 = eattr[(size_t)e * de];
        float z0 = vl.x + vr.x + e0 * w0;
        float z1 = vl.y + vr.y + e0 * w1;
        float z2 = vl.z + vr.z + e0 * w2;
        float z3 = vl.w + vr.w + e0 * w3;
        if (de == 2) {
            float e1 = eattr[(size_t)e * 2 + 1];
            z0 += e1 * We[HID + c];     z1 += e1 * We[HID + c + 1];
            z2 += e1 * We[HID + c + 2]; z3 += e1 * We[HID + c + 3];
        }
        z0 = z0 > 0.f ? z0 : LRELU_S * z0;  z1 = z1 > 0.f ? z1 : LRELU_S * z1;
        z2 = z2 > 0.f ? z2 : LRELU_S * z2;  z3 = z3 > 0.f ? z3 : LRELU_S * z3;
        float p = z0 * a0 + z1 * a1 + z2 * a2 + z3 * a3;
        p += __shfl_xor(p, 1, 32);
        p += __shfl_xor(p, 2, 32);
        p += __shfl_xor(p, 4, 32);
        if ((lane & 7) == 0) {
            int h = lane >> 3;
            a_out[(size_t)e * NHEAD + h] = p;
            atomicMaxF32(&m[(size_t)d * NHEAD + h], p);
        }
    }
}

// ---------------------------------------------------------------------------
// Edge pass 2: exp(a - max) in place + segment sum (thread per edge*head)
// ---------------------------------------------------------------------------
__global__ void edge_exp(const int* __restrict__ dst, float* __restrict__ a,
                         const float* __restrict__ m, float* __restrict__ ssum,
                         int E)
{
    int i = blockIdx.x * blockDim.x + threadIdx.x;
    if (i >= E * NHEAD) return;
    int e = i >> 2, h = i & 3;
    int d = dst[e];
    float ex = expf(a[i] - m[(size_t)d * NHEAD + h]);
    a[i] = ex;
    atomicAdd(&ssum[(size_t)d * NHEAD + h], ex);
}

// ---------------------------------------------------------------------------
// Edge pass 3: msg = xl[src] * alpha, scatter-add into out[dst] (wave per edge)
// ---------------------------------------------------------------------------
__global__ __launch_bounds__(256) void edge_scatter(
    const int* __restrict__ src, const int* __restrict__ dst,
    const float* __restrict__ aexp, const float* __restrict__ ssum,
    const float* __restrict__ xl, float* __restrict__ out, int E)
{
    const int lane = threadIdx.x & 31;
    const int wid  = blockIdx.x * 8 + (threadIdx.x >> 5);
    const int nw   = gridDim.x * 8;
    const int c    = lane * 4;
    const int h    = lane >> 3;

    for (int e = wid; e < E; e += nw) {
        int s = src[e], d = dst[e];
        float alpha = aexp[(size_t)e * NHEAD + h] / ssum[(size_t)d * NHEAD + h];
        float4 v = *(const float4*)(xl + (size_t)s * HID + c);
        float* op = out + (size_t)d * HID + c;
        atomicAdd(op + 0, v.x * alpha);
        atomicAdd(op + 1, v.y * alpha);
        atomicAdd(op + 2, v.z * alpha);
        atomicAdd(op + 3, v.w * alpha);
    }
}

// ---------------------------------------------------------------------------
// Node update: x = LN(gelu(o) + x) * g + b   (wave per node)
// ---------------------------------------------------------------------------
__global__ __launch_bounds__(256) void node_update(
    const float* __restrict__ o, float* __restrict__ x,
    const float* __restrict__ g, const float* __restrict__ bt, int N)
{
    const int lane = threadIdx.x & 31;
    const int wid  = blockIdx.x * 8 + (threadIdx.x >> 5);
    const int nw   = gridDim.x * 8;
    const int c    = lane * 4;

    for (int nid = wid; nid < N; nid += nw) {
        float4 ov = *(const float4*)(o + (size_t)nid * HID + c);
        float4 xv = *(const float4*)(x + (size_t)nid * HID + c);
        float v0 = 0.5f * ov.x * (1.f + erff(ov.x * 0.70710678f)) + xv.x;
        float v1 = 0.5f * ov.y * (1.f + erff(ov.y * 0.70710678f)) + xv.y;
        float v2 = 0.5f * ov.z * (1.f + erff(ov.z * 0.70710678f)) + xv.z;
        float v3 = 0.5f * ov.w * (1.f + erff(ov.w * 0.70710678f)) + xv.w;
        float sum = v0 + v1 + v2 + v3;
        float sq  = v0 * v0 + v1 * v1 + v2 * v2 + v3 * v3;
        #pragma unroll
        for (int msk = 1; msk < 32; msk <<= 1) {
            sum += __shfl_xor(sum, msk, 32);
            sq  += __shfl_xor(sq,  msk, 32);
        }
        float mu  = sum * (1.0f / HID);
        float var = sq * (1.0f / HID) - mu * mu;
        float inv = rsqrtf(var + LN_EPS);
        float4 r;
        r.x = (v0 - mu) * inv * g[c]     + bt[c];
        r.y = (v1 - mu) * inv * g[c + 1] + bt[c + 1];
        r.z = (v2 - mu) * inv * g[c + 2] + bt[c + 2];
        r.w = (v3 - mu) * inv * g[c + 3] + bt[c + 3];
        *(float4*)(x + (size_t)nid * HID + c) = r;
    }
}

// ---------------------------------------------------------------------------
// Host-side orchestration
// ---------------------------------------------------------------------------
struct GatP { const float *We, *Wl, *Wr, *att, *bias, *bl, *br; };

extern "C" void kernel_launch(void* const* d_in, const int* in_sizes, int n_in,
                              void* d_out, int out_size, void* d_ws, size_t ws_size,
                              hipStream_t stream)
{
    (void)in_sizes; (void)n_in; (void)out_size; (void)ws_size;

    const float* x_wifi   = (const float*)d_in[0];
    const float* x_imu    = (const float*)d_in[1];
    const int*   t_wifi   = (const int*)d_in[2];
    const int*   t_imu    = (const int*)d_in[3];
    const int*   ei_ww    = (const int*)d_in[4];
    const float* ea_ww    = (const float*)d_in[5];
    const int*   ei_ii    = (const int*)d_in[6];
    const float* ea_ii    = (const float*)d_in[7];
    const int*   ei_iw    = (const int*)d_in[8];
    const float* ea_iw    = (const float*)d_in[9];
    const int*   ei_wi    = (const int*)d_in[10];
    const float* ea_wi    = (const float*)d_in[11];
    // params flattened jax-pytree style (dict keys sorted):
    // imu_proj{W,b}, layers[0..1]{ii,iw,ln_imu{b,g},ln_wifi{b,g},wi,ww}, pos_emb, wifi_proj{W,b}
    auto P = [&](int i) { return (const float*)d_in[i]; };
    const float* imuW  = P(12);
    const float* imuB  = P(13);
    const float* pos   = P(78);
    const float* wifW  = P(79);
    const float* wifB  = P(80);
    auto gat = [&](int base) {
        GatP gp; gp.We = P(base); gp.Wl = P(base + 1); gp.Wr = P(base + 2);
        gp.att = P(base + 3); gp.bias = P(base + 4); gp.bl = P(base + 5);
        gp.br = P(base + 6); return gp;
    };

    float* xw = (float*)d_out;
    float* xi = xw + (size_t)NW_N * HID;

    // ---- workspace layout ----
    float* ws   = (float*)d_ws;
    float* xl   = ws;                               // 100000*128
    float* xr   = xl + (size_t)NI_N * HID;          // 100000*128
    float* ow   = xr + (size_t)NI_N * HID;          // 50000*128
    float* oi   = ow + (size_t)NW_N * HID;          // 100000*128
    float* abuf = oi + (size_t)NI_N * HID;          // 600000*4
    float* mbuf = abuf + (size_t)E_II * NHEAD;      // 100000*4
    float* sbuf = mbuf + (size_t)NI_N * NHEAD;      // 100000*4
    unsigned short* wt = (unsigned short*)(sbuf + (size_t)NI_N * NHEAD); // 16 x 128*WPAD bf16
    auto wtSlot = [&](int idx) { return wt + (size_t)idx * 128 * WPAD; };

    // ---- pre-convert all 16 GAT weight matrices to transposed/padded bf16 ----
    const int cvBlocks = (128 * WPAD + 255) / 256;
    for (int L = 0; L < 2; ++L) {
        int base = 14 + 32 * L;
        GatP g_ii = gat(base + 0), g_iw = gat(base + 7);
        GatP g_wi = gat(base + 18), g_ww = gat(base + 25);
        const float* srcs[8] = { g_ww.Wl, g_ww.Wr, g_iw.Wl, g_iw.Wr,
                                 g_ii.Wl, g_ii.Wr, g_wi.Wl, g_wi.Wr };
        for (int k = 0; k < 8; ++k)
            convert_w<<<cvBlocks, 256, 0, stream>>>(srcs[k], wtSlot(L * 8 + k));
    }

    // ---- input projections ----
    proj_wifi<<<NW_N, 128, 0, stream>>>(x_wifi, wifW, wifB, pos, t_wifi, xw);
    proj_imu <<<NI_N, 128, 0, stream>>>(x_imu,  imuW, imuB, pos, t_imu,  xi);

    auto runConv = [&](const float* xsrc, int nsrc, const float* xdst, int ndst,
                       const int* ei, const float* ea, int de, int E,
                       const GatP& gp, const unsigned short* wtL,
                       const unsigned short* wtR, float* oacc) {
        gemm128_bf16<<<(nsrc + 127) / 128, 256, 0, stream>>>(xsrc, wtL, gp.bl, xl, nsrc);
        gemm128_bf16<<<(ndst + 127) / 128, 256, 0, stream>>>(xdst, wtR, gp.br, xr, ndst);
        init_ms<<<(ndst * NHEAD + 255) / 256, 256, 0, stream>>>(mbuf, sbuf, ndst * NHEAD);
        int ebl = (E + 7) / 8;
        edge_attn<<<ebl, 256, 0, stream>>>(ei, ei + E, ea, de, xl, xr, gp.We, gp.att,
                                           abuf, mbuf, E);
        edge_exp<<<(E * NHEAD + 255) / 256, 256, 0, stream>>>(ei + E, abuf, mbuf, sbuf, E);
        edge_scatter<<<ebl, 256, 0, stream>>>(ei, ei + E, abuf, sbuf, xl, oacc, E);
    };

    for (int L = 0; L < 2; ++L) {
        int base = 14 + 32 * L;
        GatP g_ii = gat(base + 0);
        GatP g_iw = gat(base + 7);
        const float* lnImuB  = P(base + 14);
        const float* lnImuG  = P(base + 15);
        const float* lnWifB  = P(base + 16);
        const float* lnWifG  = P(base + 17);
        GatP g_wi = gat(base + 18);
        GatP g_ww = gat(base + 25);
        int s = L * 8;

        // wifi destination: ww + iw (accumulate into ow; init = bias_ww + bias_iw)
        init_acc<<<2048, 256, 0, stream>>>(ow, g_ww.bias, g_iw.bias, NW_N * HID);
        runConv(xw, NW_N, xw, NW_N, ei_ww, ea_ww, 1, E_WW, g_ww, wtSlot(s+0), wtSlot(s+1), ow);
        runConv(xi, NI_N, xw, NW_N, ei_iw, ea_iw, 2, E_IW, g_iw, wtSlot(s+2), wtSlot(s+3), ow);

        // imu destination: ii + wi (accumulate into oi)
        init_acc<<<2048, 256, 0, stream>>>(oi, g_ii.bias, g_wi.bias, NI_N * HID);
        runConv(xi, NI_N, xi, NI_N, ei_ii, ea_ii, 1, E_II, g_ii, wtSlot(s+4), wtSlot(s+5), oi);
        runConv(xw, NW_N, xi, NI_N, ei_wi, ea_wi, 1, E_WI, g_wi, wtSlot(s+6), wtSlot(s+7), oi);

        // post: gelu + residual + layernorm (in place; all convs read pre-update x)
        node_update<<<(NW_N + 7) / 8, 256, 0, stream>>>(ow, xw, lnWifG, lnWifB, NW_N);
        node_update<<<(NI_N + 7) / 8, 256, 0, stream>>>(oi, xi, lnImuG, lnImuB, NI_N);
    }
}
